// Entropy_21182778704536
// MI455X (gfx1250) — compile-verified, matrix operand order checked by reading, not run on hardware
//
#include <hip/hip_runtime.h>

// ---------------------------------------------------------------------------
// Problem constants (match reference)
// ---------------------------------------------------------------------------
#define NQ   4096
#define NG   16384
#define D    256
#define TOPK 50
#define TOPP 52                              // padded top-list stride (13 float4)

#define WAVES_PER_BLOCK 8
#define NG_PER_WAVE (NG / WAVES_PER_BLOCK)   // 2048 gallery cols per wave
#define CHUNKS      (NG_PER_WAVE / 16)       // 128 16-wide tiles per wave
#define QBLOCKS     (NQ / 16)                // 256 blocks
#define CAND        (WAVES_PER_BLOCK * TOPK) // 400 merged candidates / query

typedef __attribute__((ext_vector_type(16))) __bf16 v16bf;
typedef __attribute__((ext_vector_type(8)))  __bf16 v8bf;
typedef __attribute__((ext_vector_type(8)))  float  v8f;

// ---------------------------------------------------------------------------
// Wave-local LDS ordering: make one wave's DS writes visible to its own
// cross-lane DS reads (CDNA5 split counter: DScnt).
// ---------------------------------------------------------------------------
static __device__ __forceinline__ void wave_lds_sync() {
  __builtin_amdgcn_wave_barrier();
  asm volatile("s_wait_dscnt 0" ::: "memory");
  __builtin_amdgcn_wave_barrier();
}

// Pack a v16bf WMMA operand fragment for one lane.
// ISA 16-bit A-matrix 16x32 layout: lane[3:0] = row, lane[4] selects K offset
// of +8; elements 0..7 hold K = koff+0..7, elements 8..15 hold K = 16+koff+0..7.
// B (32x16, column-per-lane) mirrors this with lane = column.
static __device__ __forceinline__ v16bf load_frag(const __bf16* __restrict__ row,
                                                  int koff) {
  v8bf lo = *(const v8bf*)(row + koff);        // 16B load
  v8bf hi = *(const v8bf*)(row + 16 + koff);   // 16B load
  return __builtin_shufflevector(lo, hi,
                                 0, 1, 2, 3, 4, 5, 6, 7,
                                 8, 9, 10, 11, 12, 13, 14, 15);
}

// Load all 8 K-fragments (K=256) of one 16-row bf16 tile for this lane.
static __device__ __forceinline__ void load_frag8(v16bf b[8],
                                                  const __bf16* __restrict__ row,
                                                  int koff) {
#pragma unroll
  for (int kc = 0; kc < 8; ++kc) b[kc] = load_frag(row + kc * 32, koff);
}

// Full K=256 accumulation: 8 x v_wmma_f32_16x16x32_bf16.
static __device__ __forceinline__ v8f wmma_chain(const v16bf a[8],
                                                 const v16bf b[8]) {
  v8f c = {};
#pragma unroll
  for (int kc = 0; kc < 8; ++kc) {
    c = __builtin_amdgcn_wmma_f32_16x16x32_bf16(
            /*neg_a=*/false, a[kc], /*neg_b=*/false, b[kc],
            /*c_mod=*/(short)0, c, /*reuse_a=*/false, /*reuse_b=*/false);
  }
  return c;
}

// ---------------------------------------------------------------------------
// Kernel 1: L2-normalize each row, convert to bf16. One wave per row.
// ---------------------------------------------------------------------------
__global__ void normalize_bf16_kernel(const float* __restrict__ q,
                                      const float* __restrict__ g,
                                      __bf16* __restrict__ qbf,
                                      __bf16* __restrict__ gbf) {
  const int wave = threadIdx.x >> 5;
  const int lane = threadIdx.x & 31;
  const int row  = blockIdx.x * WAVES_PER_BLOCK + wave;

  const float* __restrict__ src;
  __bf16* __restrict__ dst;
  if (row < NQ) {
    src = q + (size_t)row * D;
    dst = qbf + (size_t)row * D;
  } else {
    src = g + (size_t)(row - NQ) * D;
    dst = gbf + (size_t)(row - NQ) * D;
  }

  // 256 floats / 32 lanes = 8 per lane (two float4 loads).
  const float4* s4 = (const float4*)(src + lane * 8);
  float4 x0 = s4[0];
  float4 x1 = s4[1];
  float f[8] = {x0.x, x0.y, x0.z, x0.w, x1.x, x1.y, x1.z, x1.w};

  float ss = 0.f;
#pragma unroll
  for (int i = 0; i < 8; ++i) ss += f[i] * f[i];
  // wave32 butterfly reduction
#pragma unroll
  for (int off = 16; off > 0; off >>= 1) ss += __shfl_xor(ss, off, 32);

  const float scale = rsqrtf(ss);
  v8bf o;
#pragma unroll
  for (int i = 0; i < 8; ++i) o[i] = (__bf16)(f[i] * scale);
  *(v8bf*)(dst + lane * 8) = o;
}

// ---------------------------------------------------------------------------
// Kernel 2: fused bf16 WMMA GEMM (Q . G^T) + streaming per-wave top-50.
// grid.x = QBLOCKS (16 query rows each); 8 waves, each owns 2048 gallery cols.
// - B fragments double-buffered in registers: chunk n+1's 16 global_load_b128
//   fly under chunk n's WMMA chain and top-k update.
// - Per-row LDS thresholds + wave-wide __any() skip the whole LDS spill/scan
//   for chunks with no top-50 candidate (most chunks after warm-up). The
//   predicate is a float max-reduction (v_max_num chain), not bool packing.
// - Scan & rescan use float4 LDS loads (clustered) instead of serial scalars.
// ---------------------------------------------------------------------------
__global__ void __launch_bounds__(256)
gemm_topk_kernel(const __bf16* __restrict__ qbf,
                 const __bf16* __restrict__ gbf,
                 float* __restrict__ topv) {
  __shared__ float s_tile[WAVES_PER_BLOCK][16][16];   // rows 64B, f4-aligned
  __shared__ float s_top[WAVES_PER_BLOCK][16][TOPP];  // 52 = 13 float4
  __shared__ float s_thr[WAVES_PER_BLOCK][16];        // per-row 50th-best

  const int w    = threadIdx.x >> 5;
  const int lane = threadIdx.x & 31;
  const int r15  = lane & 15;
  const int koff = (lane >> 4) * 8;        // K sub-offset per ISA fragment layout
  const int mbase = (lane < 16) ? 0 : 8;   // C-fragment row base for this lane
  const int qblock = blockIdx.x;

  // ---- Preload the 16x256 bf16 query fragment (reused for all 128 tiles) ----
  const __bf16* __restrict__ qrow = qbf + (size_t)(qblock * 16 + r15) * D;
  v16bf a[8];
  load_frag8(a, qrow, koff);

  // ---- Init this wave's top-50 lists, pads, and thresholds ----
  if (lane < 16) {
#pragma unroll 1
    for (int i = 0; i < TOPK; ++i) s_top[w][lane][i] = -3.0e38f;
    s_top[w][lane][TOPK]     = 3.0e38f;    // +INF pads: never the min
    s_top[w][lane][TOPK + 1] = 3.0e38f;
    s_thr[w][lane] = -3.0e38f;
  }
  wave_lds_sync();
  float vmin = -3.0e38f;
  int   imin = 0;

  // Lane's column base within this wave's gallery strip.
  const __bf16* __restrict__ gcol =
      gbf + (size_t)(w * NG_PER_WAVE + r15) * D;

  // Fold one f32 accumulator tile into the top-50 lists.
  auto tile_topk_update = [&](v8f c) {
    // Register-side filter vs per-row thresholds (two broadcast f4 loads).
    // max_i (c[i] - thr[i]) > 0  <=>  some value beats its row's 50th-best.
    float4 th0 = *(const float4*)&s_thr[w][mbase];
    float4 th1 = *(const float4*)&s_thr[w][mbase + 4];
    float dmax = fmaxf(fmaxf(fmaxf(c[0] - th0.x, c[1] - th0.y),
                             fmaxf(c[2] - th0.z, c[3] - th0.w)),
                       fmaxf(fmaxf(c[4] - th1.x, c[5] - th1.y),
                             fmaxf(c[6] - th1.z, c[7] - th1.w)));
    if (__any((int)(dmax > 0.0f))) {
      // Spill accumulator tile: lane holds rows mbase..mbase+7, col r15.
#pragma unroll
      for (int i = 0; i < 8; ++i) s_tile[w][mbase + i][r15] = c[i];
      wave_lds_sync();
      if (lane < 16) {
        const int r = lane;
        const float4* __restrict__ trow = (const float4*)&s_tile[w][r][0];
        auto consider = [&](float s) {
          if (s > vmin) {
            s_top[w][r][imin] = s;
            // Vectorized argmin rescan: 13 clustered float4 LDS loads.
            float mv = 3.0e38f;
            int   mi = 0;
            const float4* __restrict__ prow = (const float4*)&s_top[w][r][0];
#pragma unroll
            for (int b = 0; b < TOPP / 4; ++b) {
              float4 p = prow[b];
              if (p.x < mv) { mv = p.x; mi = 4 * b; }
              if (p.y < mv) { mv = p.y; mi = 4 * b + 1; }
              if (p.z < mv) { mv = p.z; mi = 4 * b + 2; }
              if (p.w < mv) { mv = p.w; mi = 4 * b + 3; }
            }
            vmin = mv;
            imin = mi;
          }
        };
#pragma unroll 1
        for (int jb = 0; jb < 4; ++jb) {
          float4 v4 = trow[jb];           // 4 candidates per LDS load
          consider(v4.x);
          consider(v4.y);
          consider(v4.z);
          consider(v4.w);
        }
        s_thr[w][r] = vmin;               // publish updated threshold
      }
      wave_lds_sync();
    }
  };

  v16bf b0[8], b1[8];
  load_frag8(b0, gcol, koff);                       // chunk 0

#pragma unroll 1
  for (int chunk = 0; chunk < CHUNKS; chunk += 2) {
    // Stage chunk+1 into b1 BEFORE consuming b0: these 16 loads overlap the
    // WMMA chain and the top-k work below.
    load_frag8(b1, gcol + (size_t)(chunk + 1) * 16 * D, koff);
    if (chunk + 2 < CHUNKS)
      __builtin_prefetch(gcol + (size_t)(chunk + 2) * 16 * D, 0, 1);

    tile_topk_update(wmma_chain(a, b0));            // chunk (even)

    if (chunk + 2 < CHUNKS)
      load_frag8(b0, gcol + (size_t)(chunk + 2) * 16 * D, koff);

    tile_topk_update(wmma_chain(a, b1));            // chunk+1 (odd)
  }

  // ---- Write this wave's partial top-50 lists: layout [query][wave][50] ----
  wave_lds_sync();
#pragma unroll 1
  for (int idx = lane; idx < 16 * TOPK; idx += 32) {
    const int rr = idx / TOPK;
    const int ii = idx % TOPK;
    topv[((size_t)(qblock * 16 + rr) * WAVES_PER_BLOCK + w) * TOPK + ii] =
        s_top[w][rr][ii];
  }
}

// ---------------------------------------------------------------------------
// Kernel 3: per query, merge 8x50 candidates -> exact top-50 -> entropy.
// ---------------------------------------------------------------------------
__global__ void merge_entropy_kernel(const float* __restrict__ topv,
                                     float* __restrict__ entb) {
  const int qIdx = blockIdx.x * blockDim.x + threadIdx.x;
  if (qIdx >= NQ) return;
  const float* __restrict__ src = topv + (size_t)qIdx * CAND;

  float t[TOPK];
#pragma unroll 1
  for (int i = 0; i < TOPK; ++i) t[i] = src[i];
  float mv = t[0];
  int   mi = 0;
#pragma unroll 1
  for (int i = 1; i < TOPK; ++i)
    if (t[i] < mv) { mv = t[i]; mi = i; }

#pragma unroll 1
  for (int j = TOPK; j < CAND; ++j) {
    float s = src[j];
    if (s > mv) {
      t[mi] = s;
      mv = t[0]; mi = 0;
#pragma unroll 1
      for (int i = 1; i < TOPK; ++i)
        if (t[i] < mv) { mv = t[i]; mi = i; }
    }
  }

  // logits = top-50 similarities; entropy of softmax (max-shifted, stable)
  float mx = t[0];
#pragma unroll 1
  for (int i = 1; i < TOPK; ++i) mx = fmaxf(mx, t[i]);
  float Z = 0.f, S = 0.f;
#pragma unroll 1
  for (int i = 0; i < TOPK; ++i) {
    float d = t[i] - mx;
    float e = __expf(d);
    Z += e;
    S += d * e;
  }
  entb[qIdx] = __logf(Z) - S / Z;
}

// ---------------------------------------------------------------------------
// Kernel 4: mean over 4096 entropies -> d_out[0].
// ---------------------------------------------------------------------------
__global__ void reduce_mean_kernel(const float* __restrict__ entb,
                                   float* __restrict__ out) {
  __shared__ float sred[256];
  float s = 0.f;
  for (int i = threadIdx.x; i < NQ; i += 256) s += entb[i];
  sred[threadIdx.x] = s;
  __syncthreads();
#pragma unroll 1
  for (int off = 128; off > 0; off >>= 1) {
    if (threadIdx.x < off) sred[threadIdx.x] += sred[threadIdx.x + off];
    __syncthreads();
  }
  if (threadIdx.x == 0) out[0] = sred[0] * (1.0f / (float)NQ);
}

// ---------------------------------------------------------------------------
// Host launcher
// ---------------------------------------------------------------------------
extern "C" void kernel_launch(void* const* d_in, const int* in_sizes, int n_in,
                              void* d_out, int out_size, void* d_ws, size_t ws_size,
                              hipStream_t stream) {
  const float* q = (const float*)d_in[0];   // [4096, 256] f32
  const float* g = (const float*)d_in[1];   // [16384, 256] f32
  float* out = (float*)d_out;               // scalar f32

  // Workspace layout (bytes):
  //   [0, 2MB)      qbf  : 4096*256 bf16
  //   [2MB, 10MB)   gbf  : 16384*256 bf16
  //   [10MB, ~16.5) topv : 4096*400 f32 partial top lists
  //   then          entb : 4096 f32 per-query entropy
  char* ws = (char*)d_ws;
  __bf16* qbf  = (__bf16*)(ws);
  __bf16* gbf  = (__bf16*)(ws + (size_t)NQ * D * sizeof(__bf16));
  float*  topv = (float*)(ws + (size_t)NQ * D * sizeof(__bf16)
                             + (size_t)NG * D * sizeof(__bf16));
  float*  entb = (float*)((char*)topv + (size_t)NQ * CAND * sizeof(float));

  normalize_bf16_kernel<<<(NQ + NG) / WAVES_PER_BLOCK, 256, 0, stream>>>(
      q, g, qbf, gbf);
  gemm_topk_kernel<<<QBLOCKS, 256, 0, stream>>>(qbf, gbf, topv);
  merge_entropy_kernel<<<NQ / 256, 256, 0, stream>>>(topv, entb);
  reduce_mean_kernel<<<1, 256, 0, stream>>>(entb, out);
}
